// EncoderRNN_77455440216599
// MI455X (gfx1250) — matrix operand order repaired and checked by reading, not compiled, not run on hardware
//
#include <hip/hip_runtime.h>

#define HD 256   // hidden size
#define BT 16    // batch tile per workgroup (WMMA M)
#define TT 128   // timesteps
#define NB 256   // utterances (batch)
#define EV 256   // embedding dim

typedef __attribute__((ext_vector_type(16))) __bf16 v16bf;
typedef __attribute__((ext_vector_type(8)))  __bf16 v8bf;
typedef __attribute__((ext_vector_type(8)))  float  v8f;

__device__ __forceinline__ unsigned short f32_to_bf16_raw(float f) {
  unsigned int u = __float_as_uint(f);
  u += 0x7FFFu + ((u >> 16) & 1u);        // round-to-nearest-even
  return (unsigned short)(u >> 16);
}
__device__ __forceinline__ float sigm(float x) { return 1.0f / (1.0f + __expf(-x)); }

// Pack [W_hh | W_ih] (fp32 row-major, rows = 4H gate rows) into bf16,
// WMMA-B fragment-contiguous: out[((ntile*KT + kt)*32 + lane)*16 + e]
//   gate row n = ntile*16 + (lane&15)
//   k          = kt*32 + (lane>>4)*8 + (e&7) + (e>>3)*16   (k<HD -> W_hh, else W_ih)
// so one wave's B-fragment for (ntile, kt) is a contiguous 1KB block.
__global__ void pack_w(const float* __restrict__ w_ih, const float* __restrict__ w_hh,
                       unsigned short* __restrict__ out, int din) {
  const int ktot = HD + din;
  const int KT = ktot >> 5;
  const long total = (long)(4 * HD) * ktot;
  for (long i = (long)blockIdx.x * blockDim.x + threadIdx.x; i < total;
       i += (long)gridDim.x * blockDim.x) {
    const int e = (int)(i & 15);
    long q = i >> 4;
    const int lane = (int)(q & 31);
    q >>= 5;
    const int kt = (int)(q % KT);
    const int ntile = (int)(q / KT);
    const int n = ntile * 16 + (lane & 15);
    const int k = kt * 32 + (lane >> 4) * 8 + (e & 7) + (e >> 3) * 16;
    const float v = (k < HD) ? w_hh[(long)n * HD + k] : w_ih[(long)n * din + (k - HD)];
    out[i] = f32_to_bf16_raw(v);
  }
}

// Embedding gather + bf16 convert, laid out [t][b][e] so WMMA-A rows are contiguous.
__global__ void embed_k(const int* __restrict__ idc, const float* __restrict__ emb,
                        unsigned short* __restrict__ xe) {
  const long total = (long)TT * NB * EV;
  for (long i = (long)blockIdx.x * blockDim.x + threadIdx.x; i < total;
       i += (long)gridDim.x * blockDim.x) {
    const int e = (int)(i & (EV - 1));
    const long tb = i >> 8;
    const int b = (int)(tb & (NB - 1));
    const int t = (int)(tb >> 8);
    const int tok = idc[b * TT + t];
    xe[i] = f32_to_bf16_raw(emb[(long)tok * EV + e]);
  }
}

// Persistent fused LSTM scan. One workgroup = 16 batch rows, 8 waves.
// Wave w owns gate columns [32w, 32w+32) for all four gates i/f/g/o, so the
// cell nonlinearity is purely per-lane VALU with c held in registers.
// grid.y = direction (0 fwd, 1 bwd). ybf (bf16 [t][b][2H]) optional; fout
// (fp32 [b][2H], last step only) optional.
__global__ void __launch_bounds__(256)
lstm_scan(const unsigned short* __restrict__ xin, int Din,
          const unsigned short* __restrict__ WF, const unsigned short* __restrict__ WB,
          const float* __restrict__ bF, const float* __restrict__ bB,
          unsigned short* __restrict__ ybf, long ytstride,
          float* __restrict__ fout, int nstepsF, int nstepsB) {
  __shared__ unsigned short h_lds[BT][HD + 8];   // +8 halfwords: kill bank conflicts

  const int dir = blockIdx.y;
  const unsigned short* __restrict__ Wc = dir ? WB : WF;
  const float* __restrict__ bias = dir ? bB : bF;
  const int nsteps = dir ? nstepsB : nstepsF;
  const int t0 = dir ? (TT - 1) : 0;
  const int tstep = dir ? -1 : 1;
  const int coloff = dir * HD;

  const int b0 = blockIdx.x * BT;
  const int wave = threadIdx.x >> 5;
  const int lane = threadIdx.x & 31;
  const int lhalf = lane >> 4;
  const int lmod = lane & 15;
  const int Ktot = HD + Din;
  const int KT = Ktot >> 5;
  const int jb = wave << 5;                       // hidden cols [jb, jb+32)

  for (int i = threadIdx.x; i < BT * (HD + 8); i += 256)
    ((unsigned short*)h_lds)[i] = 0;

  float br[4][2];
#pragma unroll
  for (int p = 0; p < 2; ++p) {
    const int j = jb + p * 16 + lmod;
    br[0][p] = bias[0 * HD + j];
    br[1][p] = bias[1 * HD + j];
    br[2][p] = bias[2 * HD + j];
    br[3][p] = bias[3 * HD + j];
  }

  const unsigned short* Wp[4][2];
#pragma unroll
  for (int g = 0; g < 4; ++g)
#pragma unroll
    for (int p = 0; p < 2; ++p) {
      const int ntile = (g << 4) + (wave << 1) + p;
      Wp[g][p] = Wc + (long)ntile * KT * 512 + lane * 16;
    }

  v8f cst[2] = {};
  __syncthreads();

  for (int s = 0; s < nsteps; ++s) {
    const int t = t0 + s * tstep;

    // Launder a zero through asm so the weight-fragment addresses are not
    // provably loop-invariant: prevents LICM from hoisting all KT*8 B-fragment
    // loads out of the scan loop (which spills ~1500 VGPRs to scratch).
    int lva = 0;
    asm volatile("" : "+s"(lva));

    v8f acc[4][2] = {};

    // K-tiles 0..7: previous hidden state (A from LDS, ISA A-frag layout)
#pragma unroll
    for (int kt = 0; kt < 8; ++kt) {
      const int ka = (kt << 5) + lhalf * 8;
      v8bf alo = *(const v8bf*)&h_lds[lmod][ka];
      v8bf ahi = *(const v8bf*)&h_lds[lmod][ka + 16];
      v16bf a = __builtin_shufflevector(alo, ahi, 0, 1, 2, 3, 4, 5, 6, 7, 8, 9,
                                        10, 11, 12, 13, 14, 15);
      // Stage all 8 B fragments first so the loads clause together.
      v16bf bws[4][2];
#pragma unroll
      for (int g = 0; g < 4; ++g)
#pragma unroll
        for (int p = 0; p < 2; ++p)
          bws[g][p] = *(const v16bf*)&Wp[g][p][(long)(kt * 512 + lva)];
#pragma unroll
      for (int g = 0; g < 4; ++g)
#pragma unroll
        for (int p = 0; p < 2; ++p)
          acc[g][p] = __builtin_amdgcn_wmma_f32_16x16x32_bf16(
              false, a, false, bws[g][p], (short)0, acc[g][p], false, false);
    }
    // K-tiles 8..KT-1: timestep input x_t (A from global bf16)
    const unsigned short* arow = xin + ((long)t * NB + (b0 + lmod)) * Din;
    for (int kt = 8; kt < KT; ++kt) {
      const int ka = ((kt - 8) << 5) + lhalf * 8;
      v8bf alo = *(const v8bf*)&arow[ka];
      v8bf ahi = *(const v8bf*)&arow[ka + 16];
      v16bf a = __builtin_shufflevector(alo, ahi, 0, 1, 2, 3, 4, 5, 6, 7, 8, 9,
                                        10, 11, 12, 13, 14, 15);
      v16bf bws[4][2];
#pragma unroll
      for (int g = 0; g < 4; ++g)
#pragma unroll
        for (int p = 0; p < 2; ++p)
          bws[g][p] = *(const v16bf*)&Wp[g][p][(long)(kt * 512 + lva)];
#pragma unroll
      for (int g = 0; g < 4; ++g)
#pragma unroll
        for (int p = 0; p < 2; ++p)
          acc[g][p] = __builtin_amdgcn_wmma_f32_16x16x32_bf16(
              false, a, false, bws[g][p], (short)0, acc[g][p], false, false);
    }

    __syncthreads();  // all h_lds reads for this step done

    // LSTM cell: C/D layout => lane holds (M=r+8*lhalf, N=lmod); i/f/g/o for
    // one (b,j) share lane+slot across the four accumulators.
#pragma unroll
    for (int p = 0; p < 2; ++p) {
      const int j = jb + p * 16 + lmod;
#pragma unroll
      for (int r = 0; r < 8; ++r) {
        const float iv = sigm(acc[0][p][r] + br[0][p]);
        const float fv = sigm(acc[1][p][r] + br[1][p]);
        const float gv = tanhf(acc[2][p][r] + br[2][p]);
        const float ov = sigm(acc[3][p][r] + br[3][p]);
        const float cv = fv * cst[p][r] + iv * gv;
        cst[p][r] = cv;
        const float hv = ov * tanhf(cv);
        const int m = r + 8 * lhalf;
        const unsigned short hb = f32_to_bf16_raw(hv);
        h_lds[m][j] = hb;
        if (ybf)
          ybf[(long)t * ytstride + (long)(b0 + m) * (2 * HD) + coloff + j] = hb;
        if (fout && s == nsteps - 1)
          fout[(long)(b0 + m) * (2 * HD) + coloff + j] = hv;
      }
    }
    __syncthreads();  // new h visible before next step's reads
  }
}

extern "C" void kernel_launch(void* const* d_in, const int* in_sizes, int n_in,
                              void* d_out, int out_size, void* d_ws, size_t ws_size,
                              hipStream_t stream) {
  (void)in_sizes; (void)n_in; (void)out_size; (void)ws_size;
  const int* idc = (const int*)d_in[0];
  const float* emb = (const float*)d_in[1];
  const float* w_ih0f = (const float*)d_in[2];
  const float* w_hh0f = (const float*)d_in[3];
  const float* b0f = (const float*)d_in[4];
  const float* w_ih0b = (const float*)d_in[5];
  const float* w_hh0b = (const float*)d_in[6];
  const float* b0b = (const float*)d_in[7];
  const float* w_ih1f = (const float*)d_in[8];
  const float* w_hh1f = (const float*)d_in[9];
  const float* b1f = (const float*)d_in[10];
  const float* w_ih1b = (const float*)d_in[11];
  const float* w_hh1b = (const float*)d_in[12];
  const float* b1b = (const float*)d_in[13];

  char* ws = (char*)d_ws;
  size_t off = 0;
  auto wsalloc = [&](size_t bytes) -> void* {
    off = (off + 255) & ~(size_t)255;
    void* p = ws + off;
    off += bytes;
    return p;
  };
  unsigned short* xe = (unsigned short*)wsalloc((size_t)TT * NB * EV * 2);          // 16 MB
  unsigned short* y0 = (unsigned short*)wsalloc((size_t)TT * NB * 2 * HD * 2);      // 32 MB
  unsigned short* Wc0f = (unsigned short*)wsalloc((size_t)4 * HD * (HD + EV) * 2);
  unsigned short* Wc0b = (unsigned short*)wsalloc((size_t)4 * HD * (HD + EV) * 2);
  unsigned short* Wc1f = (unsigned short*)wsalloc((size_t)4 * HD * (3 * HD) * 2);
  unsigned short* Wc1b = (unsigned short*)wsalloc((size_t)4 * HD * (3 * HD) * 2);

  pack_w<<<1024, 256, 0, stream>>>(w_ih0f, w_hh0f, Wc0f, EV);
  pack_w<<<1024, 256, 0, stream>>>(w_ih0b, w_hh0b, Wc0b, EV);
  pack_w<<<1024, 256, 0, stream>>>(w_ih1f, w_hh1f, Wc1f, 2 * HD);
  pack_w<<<1024, 256, 0, stream>>>(w_ih1b, w_hh1b, Wc1b, 2 * HD);
  embed_k<<<2048, 256, 0, stream>>>(idc, emb, xe);

  // Layer 0: both directions, full 128-step scans, y0 = [t][b][2H] bf16
  lstm_scan<<<dim3(NB / BT, 2), 256, 0, stream>>>(
      xe, EV, Wc0f, Wc0b, b0f, b0b, y0, (long)NB * 2 * HD, (float*)nullptr, TT, TT);

  // Layer 1: fwd full scan (keep last h); bwd needs only ONE step, since
  // y_bwd[T-1] (after re-reversal) is the backward scan's first step on x[T-1].
  lstm_scan<<<dim3(NB / BT, 2), 256, 0, stream>>>(
      y0, 2 * HD, Wc1f, Wc1b, b1f, b1b, (unsigned short*)nullptr, 0,
      (float*)d_out, TT, 1);
}
